// TriangleMultiplicativeUpdate_55688545960633
// MI455X (gfx1250) — matrix operand order repaired and checked by reading, not compiled, not run on hardware
//
#include <hip/hip_runtime.h>
#include <stdint.h>

typedef unsigned short u16;
typedef unsigned int   u32;
typedef __attribute__((ext_vector_type(16))) __bf16 v16bf;
typedef __attribute__((ext_vector_type(8)))  float  v8f;
typedef __attribute__((ext_vector_type(4)))  u32    v4u;
typedef __attribute__((ext_vector_type(8)))  int    v8i;
typedef __attribute__((ext_vector_type(4)))  int    v4i;

#define S  512
#define CZ 128
#define CH 128

__device__ __forceinline__ u16 f2bf(float x) {
  union { float f; unsigned u; } v; v.f = x;
  unsigned r = v.u + 0x7FFFu + ((v.u >> 16) & 1u);
  return (u16)(r >> 16);
}
__device__ __forceinline__ float bf2f(u16 h) {
  union { unsigned u; float f; } v; v.u = ((unsigned)h) << 16;
  return v.f;
}
__device__ __forceinline__ v8f wmma_bf16(v16bf a, v16bf b, v8f c) {
  // D = A(16x32 bf16) * B(32x16 bf16) + C(16x16 f32)
  return __builtin_amdgcn_wmma_f32_16x16x32_bf16(false, a, false, b, (short)0, c, false, false);
}

// A fragment from a row-major 16 x 128 bf16 tile in LDS (padded stride).
// Lane holds row M = lane&15; K = kc*32 + [8*(lane>>4) .. +8) and [+16 .. +24).
__device__ __forceinline__ v16bf frag_a_lds(const u16* t, int lane, int kc, int stride) {
  int row = lane & 15;
  int kb  = kc * 32 + ((lane >> 4) << 3);
  union { v16bf v; uint4 q[2]; } u;
  const u16* p = t + row * stride + kb;
  u.q[0] = *(const uint4*)(p);
  u.q[1] = *(const uint4*)(p + 16);
  return u.v;
}
// B fragment from global Wt[n][k] (row-major, 128 wide, bf16 = W transposed).
// Lane holds column N = lane&15; K = kc*32 + [8*(lane>>4) .. +8) and [+16 .. +24).
__device__ __forceinline__ v16bf frag_b_wt(const u16* Wt, int lane, int nt, int kc) {
  int n  = nt * 16 + (lane & 15);
  int kb = kc * 32 + ((lane >> 4) << 3);
  union { v16bf v; uint4 q[2]; } u;
  const u16* p = Wt + n * 128 + kb;
  u.q[0] = *(const uint4*)(p);
  u.q[1] = *(const uint4*)(p + 16);
  return u.v;
}

// ---------------------------------------------------------------------------
// TDM: issue a 2D tile load (128 rows x 32 bf16, row stride 512 bf16) from
// global into LDS with hardware padding: 64B of data then 16B pad -> 80B LDS
// row stride (= 40 u16, bank-conflict-free for the WMMA fragment reads).
// D# packing per CDNA5 ISA.
// Toolchain uses the 6-arg builtin: (g0 u32x4, g1 i32x8, g2 i32x4, g3 i32x4,
// extra i32x8, cpol i32); unused groups zero-filled.
// ---------------------------------------------------------------------------
__device__ __forceinline__ void tdm_load_tile(u32 lds_off, unsigned long long gaddr) {
  v4u g0;
  g0[0] = 1u;                                        // count=1, user descriptor
  g0[1] = lds_off;                                   // lds_addr (bytes)
  g0[2] = (u32)(gaddr & 0xFFFFFFFFull);              // global_addr[31:0]
  g0[3] = (u32)((gaddr >> 32) & 0x1FFFFFFull)        // global_addr[56:32]
        | (2u << 30);                                // type = 2 ("image")
  v8i g1;
  g1[0] = (int)((1u << 16)      // data_size = 1 -> 2 bytes
              | (1u << 20)      // pad_enable
              | (3u << 22)      // pad_interval code 3 -> every 16 DWORDs (64B)
              | (3u << 25));    // pad_amount  code 3 -> 4 DWORDs (16B)
  g1[1] = (int)(512u << 16);    // tensor_dim0 = 512 (bits 79:48)
  g1[2] = (int)(512u << 16);    // tensor_dim0 hi=0 | tensor_dim1 lo (=512)
  g1[3] = (int)(32u << 16);     // tensor_dim1 hi=0 | tile_dim0 = 32
  g1[4] = 128;                  // tile_dim1 = 128, tile_dim2 = 0 (2D)
  g1[5] = 512;                  // tensor_dim0_stride = 512 elements
  g1[6] = 0;                    // stride hi | tensor_dim1_stride lo
  g1[7] = 4;                    // tensor_dim1_stride hi (512*512 >> 16 = 4)
  v4i z4 = {0, 0, 0, 0};        // groups 2/3 unused for 2D tiles
  v8i z8 = {0, 0, 0, 0, 0, 0, 0, 0};
  __builtin_amdgcn_tensor_load_to_lds(g0, g1, z4, z4, z8, 0);
}

// ---------------------------------------------------------------------------
// k0: convert & transpose the four 128x128 weight matrices to bf16 Wt[n][k]
// ---------------------------------------------------------------------------
__global__ __launch_bounds__(128) void k0_prep(
    const float* __restrict__ Wa, const float* __restrict__ Wb,
    const float* __restrict__ Wg, const float* __restrict__ Wo,
    u16* __restrict__ WtA, u16* __restrict__ WtB,
    u16* __restrict__ WtG, u16* __restrict__ WtO)
{
  int m = blockIdx.y;            // which matrix
  int n = blockIdx.x;            // output row of Wt (= column of W)
  int k = threadIdx.x;           // input channel
  const float* W = (m == 0) ? Wa : (m == 1) ? Wb : (m == 2) ? Wg : Wo;
  u16*        Wt = (m == 0) ? WtA : (m == 1) ? WtB : (m == 2) ? WtG : WtO;
  Wt[n * 128 + k] = f2bf(W[k * 128 + n]);
}

// ---------------------------------------------------------------------------
// k1: fused LayerNorm + projections a, b, gate.
// Block = 16 consecutive (i,k) rows (same i), 128 threads = 4 waves.
// a/b stored channel-major bf16 a_t[c][i][k]; gate sigmoided, row-major bf16.
// ---------------------------------------------------------------------------
__global__ __launch_bounds__(128) void k1_ln_proj(
    const float* __restrict__ z, const float* __restrict__ ln_w, const float* __restrict__ ln_b,
    const float* __restrict__ ba, const float* __restrict__ bb, const float* __restrict__ bg,
    const u16* __restrict__ WtA, const u16* __restrict__ WtB, const u16* __restrict__ WtG,
    u16* __restrict__ a_t, u16* __restrict__ b_t, u16* __restrict__ g_out)
{
  __shared__ float zt[16 * 132];   // fp32 rows, padded stride (bank-conflict-free)
  __shared__ u16   zn[16 * 136];   // normalized bf16 rows, padded stride
  const int tid = threadIdx.x;
  const int lane = tid & 31, wave = tid >> 5;
  const int r0 = blockIdx.x * 16;          // first linear (i,k) row
  const int i  = r0 >> 9;                  // i index (S=512)
  const int k0 = r0 & 511;                 // k base

  // Keep the weight tables hot in L2/WGP$.
  __builtin_prefetch(WtA + (tid << 7), 0, 3);
  __builtin_prefetch(WtB + (tid << 7), 0, 3);
  __builtin_prefetch(WtG + (tid << 7), 0, 3);

  // Load 16 rows x 128 ch of z (float4, fully coalesced).
  const float4* zsrc = (const float4*)(z + (size_t)r0 * CZ);
  #pragma unroll
  for (int it = 0; it < 4; ++it) {
    int idx = tid + it * 128;              // 0..511 float4s
    int row = idx >> 5, c4 = idx & 31;
    float4 v = zsrc[idx];
    float* d = &zt[row * 132 + c4 * 4];
    d[0] = v.x; d[1] = v.y; d[2] = v.z; d[3] = v.w;
  }
  __syncthreads();

  // LayerNorm: each wave handles 4 rows, 128 elems across 32 lanes.
  #pragma unroll
  for (int rr = 0; rr < 4; ++rr) {
    int row = wave * 4 + rr;
    float x[4], s = 0.f, s2 = 0.f;
    #pragma unroll
    for (int j = 0; j < 4; ++j) {
      x[j] = zt[row * 132 + lane + 32 * j];
      s += x[j]; s2 += x[j] * x[j];
    }
    #pragma unroll
    for (int off = 16; off > 0; off >>= 1) {
      s  += __shfl_xor(s,  off, 32);
      s2 += __shfl_xor(s2, off, 32);
    }
    float mu  = s * (1.0f / 128.0f);
    float inv = rsqrtf(s2 * (1.0f / 128.0f) - mu * mu + 1e-5f);
    #pragma unroll
    for (int j = 0; j < 4; ++j) {
      int c = lane + 32 * j;
      zn[row * 136 + c] = f2bf((x[j] - mu) * inv * ln_w[c] + ln_b[c]);
    }
  }
  __syncthreads();

  // A fragments (16 rows x 128 k), shared across all three projections.
  v16bf afrag[4];
  #pragma unroll
  for (int kc = 0; kc < 4; ++kc) afrag[kc] = frag_a_lds(zn, lane, kc, 136);

  #pragma unroll
  for (int m = 0; m < 3; ++m) {
    const u16*  Wt   = (m == 0) ? WtA : (m == 1) ? WtB : WtG;
    const float* bias = (m == 0) ? ba : (m == 1) ? bb : bg;
    u16* dstm = (m == 0) ? a_t : b_t;
    #pragma unroll
    for (int t2 = 0; t2 < 2; ++t2) {
      int nt = wave * 2 + t2;                       // 8 N-tiles over 4 waves
      v8f acc = {0.f, 0.f, 0.f, 0.f, 0.f, 0.f, 0.f, 0.f};
      #pragma unroll
      for (int kc = 0; kc < 4; ++kc)
        acc = wmma_bf16(afrag[kc], frag_b_wt(Wt, lane, nt, kc), acc);

      int c = nt * 16 + (lane & 15);
      float bv = bias[c];
      if (m < 2) {
        // Transposed store: D-layout VGPRs are 8 consecutive rows (=k) at
        // fixed column (=c) -> one contiguous 16B store per lane.
        u16 o[8];
        #pragma unroll
        for (int e = 0; e < 8; ++e) o[e] = f2bf(acc[e] + bv);
        u16* d = dstm + (size_t)c * (size_t)(S * S) + (size_t)i * S
                      + k0 + ((lane >> 4) << 3);
        *(uint4*)d = *(const uint4*)o;
      } else {
        // Gate: sigmoid, row-major [pos][c] bf16.
        #pragma unroll
        for (int e = 0; e < 8; ++e) {
          int M = e + ((lane >> 4) << 3);
          float v = acc[e] + bv;
          g_out[(size_t)(r0 + M) * CZ + c] = f2bf(1.0f / (1.0f + __expf(-v)));
        }
      }
    }
  }
}

// ---------------------------------------------------------------------------
// k2: triangle contraction. Per channel c: C = A * B^T, A=a_t[c] (i x k),
// B rows = b_t[c] (j x k). 128x128 block tile; K staged through double-
// buffered LDS by the Tensor Data Mover (wave 0 issues, TENSORcnt + barrier
// publishes). TDM pad fields create the conflict-free 40-u16 LDS row stride.
// ---------------------------------------------------------------------------
__global__ __launch_bounds__(256) void k2_outer(
    const u16* __restrict__ a_t, const u16* __restrict__ b_t,
    float* __restrict__ ab_t)
{
  __shared__ u16 As[2][128 * 40];   // 128 rows x 32 k, stride 40 u16 (80B)
  __shared__ u16 Bs[2][128 * 40];
  const int tid = threadIdx.x, lane = tid & 31, wave = tid >> 5;
  const int c  = blockIdx.z;
  const int i0 = blockIdx.y * 128, j0 = blockIdx.x * 128;
  const size_t cb = (size_t)c * (size_t)(S * S);

  // Byte addresses of the (i0,0)/(j0,0) tile corners for this channel.
  const unsigned long long gA =
      (unsigned long long)(size_t)(a_t + cb + (size_t)i0 * S);
  const unsigned long long gB =
      (unsigned long long)(size_t)(b_t + cb + (size_t)j0 * S);
  const u32 ldsA[2] = { (u32)(size_t)(void*)As[0], (u32)(size_t)(void*)As[1] };
  const u32 ldsB[2] = { (u32)(size_t)(void*)Bs[0], (u32)(size_t)(void*)Bs[1] };

  v8f acc[8];
  const v8f vzero = {0.f, 0.f, 0.f, 0.f, 0.f, 0.f, 0.f, 0.f};
  #pragma unroll
  for (int t = 0; t < 8; ++t) acc[t] = vzero;

  // Prologue: stage K-chunk 0 into buffer 0.
  if (wave == 0) {
    tdm_load_tile(ldsA[0], gA);
    tdm_load_tile(ldsB[0], gB);
  }

  for (int kk = 0; kk < 16; ++kk) {
    const int cur = kk & 1;
    if (wave == 0) {
      if (kk + 1 < 16) {
        // Prefetch next K-chunk into the buffer freed at the end of kk-1.
        unsigned long long koff = (unsigned long long)(kk + 1) * 32ull * 2ull;
        tdm_load_tile(ldsA[cur ^ 1], gA + koff);
        tdm_load_tile(ldsB[cur ^ 1], gB + koff);
        __builtin_amdgcn_s_wait_tensorcnt(2);  // retire cur-buffer pair
      } else {
        __builtin_amdgcn_s_wait_tensorcnt(0);
      }
    }
    __syncthreads();   // publish cur buffer to all 8 waves

    // A fragment: wave's 16-row slab.
    union { v16bf v; uint4 q[2]; } ua;
    {
      int row = 16 * wave + (lane & 15);
      int kb  = (lane >> 4) << 3;
      const u16* p = &As[cur][row * 40 + kb];
      ua.q[0] = *(const uint4*)(p);
      ua.q[1] = *(const uint4*)(p + 16);
    }
    #pragma unroll
    for (int t = 0; t < 8; ++t) {
      union { v16bf v; uint4 q[2]; } ub;
      int jr = 16 * t + (lane & 15);
      int kb = (lane >> 4) << 3;
      const u16* p = &Bs[cur][jr * 40 + kb];
      ub.q[0] = *(const uint4*)(p);
      ub.q[1] = *(const uint4*)(p + 16);
      acc[t] = wmma_bf16(ua.v, ub.v, acc[t]);
    }
    __syncthreads();   // all reads of cur done before TDM reuses it
  }

  // Store ab_t[c][i][j] fp32, coalesced (lanes 0-15 consecutive j).
  float* dst = ab_t + cb;
  #pragma unroll
  for (int t = 0; t < 8; ++t) {
    int j = j0 + 16 * t + (lane & 15);
    #pragma unroll
    for (int e = 0; e < 8; ++e) {
      int i = i0 + 16 * wave + e + ((lane >> 4) << 3);
      dst[(size_t)i * S + j] = acc[t][e];
    }
  }
}

// ---------------------------------------------------------------------------
// k3: LN over c, project with Wout, add bias, multiply by gate, store fp32.
// Block = 16 linear (i,j) positions, 128 threads = 4 waves.
// ---------------------------------------------------------------------------
__global__ __launch_bounds__(128) void k3_out(
    const float* __restrict__ ab_t, const u16* __restrict__ g_in,
    const float* __restrict__ lno_w, const float* __restrict__ lno_b,
    const u16* __restrict__ WtO, const float* __restrict__ bout,
    float* __restrict__ out)
{
  __shared__ float at[16 * 132];
  __shared__ u16   zn[16 * 136];
  __shared__ u16   gt[16 * 136];
  const int tid = threadIdx.x, lane = tid & 31, wave = tid >> 5;
  const size_t p0 = (size_t)blockIdx.x * 16;

  __builtin_prefetch(WtO + (tid << 7), 0, 3);

  // Gather ab_t[c][p0..p0+15] (64B contiguous per channel) -> at[p][c].
  {
    const float* src = ab_t + (size_t)tid * (size_t)(S * S) + p0;
    #pragma unroll
    for (int q = 0; q < 4; ++q) {
      float4 v = *(const float4*)(src + q * 4);
      at[(q * 4 + 0) * 132 + tid] = v.x;
      at[(q * 4 + 1) * 132 + tid] = v.y;
      at[(q * 4 + 2) * 132 + tid] = v.z;
      at[(q * 4 + 3) * 132 + tid] = v.w;
    }
    // Gate tile: 16 x 128 bf16, contiguous in global.
    const uint4* gs = (const uint4*)(g_in + p0 * CZ);
    int p = tid >> 3, c8 = (tid & 7) * 16;
    uint4* gd = (uint4*)(&gt[p * 136 + c8]);
    gd[0] = gs[tid * 2 + 0];
    gd[1] = gs[tid * 2 + 1];
  }
  __syncthreads();

  // LayerNorm over c_hid.
  #pragma unroll
  for (int rr = 0; rr < 4; ++rr) {
    int row = wave * 4 + rr;
    float x[4], s = 0.f, s2 = 0.f;
    #pragma unroll
    for (int j = 0; j < 4; ++j) {
      x[j] = at[row * 132 + lane + 32 * j];
      s += x[j]; s2 += x[j] * x[j];
    }
    #pragma unroll
    for (int off = 16; off > 0; off >>= 1) {
      s  += __shfl_xor(s,  off, 32);
      s2 += __shfl_xor(s2, off, 32);
    }
    float mu  = s * (1.0f / 128.0f);
    float inv = rsqrtf(s2 * (1.0f / 128.0f) - mu * mu + 1e-5f);
    #pragma unroll
    for (int j = 0; j < 4; ++j) {
      int c = lane + 32 * j;
      zn[row * 136 + c] = f2bf((x[j] - mu) * inv * lno_w[c] + lno_b[c]);
    }
  }
  __syncthreads();

  v16bf afrag[4];
  #pragma unroll
  for (int kc = 0; kc < 4; ++kc) afrag[kc] = frag_a_lds(zn, lane, kc, 136);

  #pragma unroll
  for (int t2 = 0; t2 < 2; ++t2) {
    int nt = wave * 2 + t2;
    v8f acc = {0.f, 0.f, 0.f, 0.f, 0.f, 0.f, 0.f, 0.f};
    #pragma unroll
    for (int kc = 0; kc < 4; ++kc)
      acc = wmma_bf16(afrag[kc], frag_b_wt(WtO, lane, nt, kc), acc);

    int zc = nt * 16 + (lane & 15);
    float bv = bout[zc];
    #pragma unroll
    for (int e = 0; e < 8; ++e) {
      int M = e + ((lane >> 4) << 3);
      float gate = bf2f(gt[M * 136 + zc]);
      out[(p0 + M) * CZ + zc] = gate * (acc[e] + bv);
    }
  }
}

// ---------------------------------------------------------------------------
// Workspace layout (bytes):
//   [0,      32K)  WtA bf16    [32K,  64K)  WtB bf16
//   [64K,   96K)   WtG bf16    [96K, 128K)  WtO bf16
//   [128K, 128K+64M)        a_t  bf16 [c][i][k]
//   [128K+64M, 128K+128M)   b_t  bf16 [c][j][k]
//   [128K+128M, 128K+192M)  gate bf16 [p][c] (sigmoided)
//   [128K+192M, 128K+320M)  ab_t fp32 [c][i][j]
// ---------------------------------------------------------------------------
extern "C" void kernel_launch(void* const* d_in, const int* in_sizes, int n_in,
                              void* d_out, int out_size, void* d_ws, size_t ws_size,
                              hipStream_t stream) {
  const float* z     = (const float*)d_in[0];
  const float* ln_w  = (const float*)d_in[1];
  const float* ln_b  = (const float*)d_in[2];
  const float* Wa    = (const float*)d_in[3];
  const float* ba    = (const float*)d_in[4];
  const float* Wb    = (const float*)d_in[5];
  const float* bb    = (const float*)d_in[6];
  const float* Wg    = (const float*)d_in[7];
  const float* bg    = (const float*)d_in[8];
  const float* lno_w = (const float*)d_in[9];
  const float* lno_b = (const float*)d_in[10];
  const float* Wout  = (const float*)d_in[11];
  const float* bout  = (const float*)d_in[12];
  float* out = (float*)d_out;

  char* ws = (char*)d_ws;
  const size_t MB64 = 67108864ull;
  u16*   WtA  = (u16*)(ws);
  u16*   WtB  = (u16*)(ws + 32768);
  u16*   WtG  = (u16*)(ws + 65536);
  u16*   WtO  = (u16*)(ws + 98304);
  u16*   a_t  = (u16*)(ws + 131072);
  u16*   b_t  = (u16*)(ws + 131072 + MB64);
  u16*   gbuf = (u16*)(ws + 131072 + 2 * MB64);
  float* ab_t = (float*)(ws + 131072 + 3 * MB64);

  hipLaunchKernelGGL(k0_prep, dim3(128, 4), dim3(128), 0, stream,
                     Wa, Wb, Wg, Wout, WtA, WtB, WtG, WtO);
  hipLaunchKernelGGL(k1_ln_proj, dim3((S * S) / 16), dim3(128), 0, stream,
                     z, ln_w, ln_b, ba, bb, bg, WtA, WtB, WtG, a_t, b_t, gbuf);
  hipLaunchKernelGGL(k2_outer, dim3(4, 4, 128), dim3(256), 0, stream,
                     a_t, b_t, ab_t);
  hipLaunchKernelGGL(k3_out, dim3((S * S) / 16), dim3(128), 0, stream,
                     ab_t, gbuf, lno_w, lno_b, WtO, bout, out);
}
